// LowRank_78864189489811
// MI455X (gfx1250) — matrix operand order repaired and checked by reading, not compiled, not run on hardware
//
#include <hip/hip_runtime.h>
#include <hip/hip_bf16.h>
#include <math.h>

typedef __bf16 bf16_t;
typedef __attribute__((ext_vector_type(16))) __bf16 v16bf;
typedef __attribute__((ext_vector_type(8)))  __bf16 v8bf;
typedef __attribute__((ext_vector_type(8)))  float  v8f;

#define DDIM 1024
#define HH   8
#define HDIM 128
#define MM1  64
#define CELU_ALPHA 1.3f
#define GN_EPS 1e-5f

#ifndef __has_builtin
#define __has_builtin(x) 0
#endif
#if __has_builtin(__builtin_amdgcn_global_load_async_to_lds_b128) && \
    __has_builtin(__builtin_amdgcn_s_wait_asynccnt)
#define HAS_ASYNC 1
#else
#define HAS_ASYNC 0
#endif

#if HAS_ASYNC
typedef int v4i_t __attribute__((vector_size(4 * sizeof(int))));
typedef __attribute__((address_space(1))) v4i_t* as1_v4i;
typedef __attribute__((address_space(3))) v4i_t* as3_v4i;
#endif

// ---------------------------------------------------------------------------
// fp32 weight (K x N row-major) -> bf16 transposed (N x K): WMMA B fragments
// become two contiguous 16-byte loads per lane.
// ---------------------------------------------------------------------------
__global__ void convert_transpose(const float* __restrict__ W,
                                  bf16_t* __restrict__ Wt, int K, int N) {
  int idx = blockIdx.x * blockDim.x + threadIdx.x;
  if (idx >= K * N) return;
  int k = idx / N, n = idx % N;               // read coalesced along n
  Wt[(size_t)n * K + k] = (bf16_t)W[idx];
}

// ---------------------------------------------------------------------------
// Fused branch: Y = group_norm(celu(X @ W + b)), 16 rows per workgroup.
// 8 waves; wave w owns columns [w*128, w*128+128) == one group-norm group.
// A tile double-buffered in LDS (fp32), filled via async global->LDS copies
// when available; B fragments batched per k-step for clause formation.
// ---------------------------------------------------------------------------
__global__ __launch_bounds__(256)
void branch_kernel(const float* __restrict__ X, const bf16_t* __restrict__ Wt,
                   const float* __restrict__ bias, const float* __restrict__ gamma,
                   const float* __restrict__ beta, float* __restrict__ outF,
                   bf16_t* __restrict__ outB)
{
  __shared__ __align__(16) float As[2][16][32];   // double-buffered A tile
  __shared__ float Rs[16][1024];                  // celu(x@W+b) result tile
  __shared__ float muS[16][8], rsS[16][8];        // group stats

  const int tid  = threadIdx.x;
  const int lane = tid & 31;
  const int wave = tid >> 5;
  const int ml   = lane & 15;     // row (A) / col (B) within 16
  const int kg   = lane >> 4;     // lane-half selects K sub-chunk
  const size_t row0 = (size_t)blockIdx.x * 16;

  v8f acc[8] = {};                // 8 column tiles of 16x16 f32

  auto fillA = [&](int kk, int buf) {
#if HAS_ASYNC
    if (tid < 128) {
      int r = tid >> 3, c4 = (tid & 7) * 4;     // 16 rows x 8 x 16B chunks
      const float* gp = &X[(row0 + r) * DDIM + kk + c4];
      float* lp = &As[buf][r][c4];
      __builtin_amdgcn_global_load_async_to_lds_b128(
          (as1_v4i)(void*)gp, (as3_v4i)(void*)lp, 0, 0);
    }
#else
    int f = tid;
    int r = f >> 5, c = f & 31;
    As[buf][r][c] = X[(row0 + r) * DDIM + kk + c];
    f += 256; r = f >> 5; c = f & 31;
    As[buf][r][c] = X[(row0 + r) * DDIM + kk + c];
#endif
  };

  fillA(0, 0);
#if HAS_ASYNC
  __builtin_amdgcn_s_wait_asynccnt(0);
#endif
  __syncthreads();

  for (int kk = 0; kk < DDIM; kk += 32) {
    const int cur = (kk >> 5) & 1;
    if (kk + 32 < DDIM) fillA(kk + 32, cur ^ 1);   // overlap next fill
    if (kk + 64 < DDIM)
      __builtin_prefetch(&X[(row0 + (tid >> 4)) * DDIM + kk + 64], 0, 1);

    // A fragment: lane ml = row; K = kk+kg*8+{0..7}, kk+16+kg*8+{0..7}
    const float4* arow = (const float4*)&As[cur][ml][0];
    float4 a0 = arow[kg * 2], a1 = arow[kg * 2 + 1];
    float4 a2 = arow[4 + kg * 2], a3 = arow[5 + kg * 2];
    v16bf a;
    a[0]=(bf16_t)a0.x; a[1]=(bf16_t)a0.y; a[2]=(bf16_t)a0.z; a[3]=(bf16_t)a0.w;
    a[4]=(bf16_t)a1.x; a[5]=(bf16_t)a1.y; a[6]=(bf16_t)a1.z; a[7]=(bf16_t)a1.w;
    a[8]=(bf16_t)a2.x; a[9]=(bf16_t)a2.y; a[10]=(bf16_t)a2.z; a[11]=(bf16_t)a2.w;
    a[12]=(bf16_t)a3.x; a[13]=(bf16_t)a3.y; a[14]=(bf16_t)a3.z; a[15]=(bf16_t)a3.w;

    // batched B loads: 16 x global_load_b128 clause, then 8 WMMAs
    const bf16_t* bbase = Wt + (size_t)(wave * 128 + ml) * DDIM + kk;
    v8bf blo[8], bhi[8];
#pragma unroll
    for (int t = 0; t < 8; ++t) {
      const v8bf* brow = (const v8bf*)(bbase + (size_t)t * 16 * DDIM);
      blo[t] = brow[kg];
      bhi[t] = brow[2 + kg];
    }
#pragma unroll
    for (int t = 0; t < 8; ++t) {
      v16bf bfrag;
#pragma unroll
      for (int i = 0; i < 8; ++i) { bfrag[i] = blo[t][i]; bfrag[i + 8] = bhi[t][i]; }
      acc[t] = __builtin_amdgcn_wmma_f32_16x16x32_bf16(
          false, a, false, bfrag, (short)0, acc[t], false, false);
    }
#if HAS_ASYNC
    __builtin_amdgcn_s_wait_asynccnt(0);
#endif
    __syncthreads();
  }

  // bias + celu, scatter to LDS tile (C/D layout: row = jj + kg*8)
#pragma unroll
  for (int t = 0; t < 8; ++t) {
#pragma unroll
    for (int jj = 0; jj < 8; ++jj) {
      int m = jj + kg * 8;
      int n = wave * 128 + t * 16 + ml;
      float x = acc[t][jj] + bias[n];
      Rs[m][n] = x > 0.f ? x : CELU_ALPHA * expm1f(x / CELU_ALPHA);
    }
  }
  __syncthreads();

  // group-norm stats: 16 rows x 8 groups of 128
  if (tid < 128) {
    int m = tid >> 3, g = tid & 7;
    float s = 0.f, sq = 0.f;
    for (int j = 0; j < 128; ++j) {
      float v = Rs[m][g * 128 + j];
      s += v; sq += v * v;
    }
    float mean = s * (1.f / 128.f);
    float var  = sq * (1.f / 128.f) - mean * mean;
    muS[m][g] = mean;
    rsS[m][g] = rsqrtf(var + GN_EPS);
  }
  __syncthreads();

  // normalize + affine + write out (fp32 and/or bf16)
  for (int i = 0; i < 64; ++i) {
    int flat = tid + i * 256;
    int m = flat >> 10, n = flat & 1023, g = n >> 7;
    float y = (Rs[m][n] - muS[m][g]) * rsS[m][g] * gamma[n] + beta[n];
    size_t o = (row0 + m) * DDIM + n;
    if (outF) outF[o] = y;
    if (outB) outB[o] = (bf16_t)y;
  }
}

// ---------------------------------------------------------------------------
// Fused attention tail: one workgroup per (b, h).
//  am = relu((q .* k) @ Wb + bb) via bf16 WMMA (K=128, 4 steps, Wb fragments
//  hoisted into VGPRs), s[m]/pool[n] accumulated with LDS float atomics,
//  masked softmax, channel gate, pooled_v2 via vectorized bf16 loads.
// ---------------------------------------------------------------------------
__global__ __launch_bounds__(256)
void attn_kernel(const float* __restrict__ qF, const bf16_t* __restrict__ kB,
                 const float* __restrict__ v1F, const bf16_t* __restrict__ v2B,
                 const float* __restrict__ mask,
                 const bf16_t* __restrict__ Wbt, const float* __restrict__ bb,
                 const float* __restrict__ Wl, const float* __restrict__ bl,
                 const float* __restrict__ Wl2, const float* __restrict__ bl2,
                 float* __restrict__ out)
{
  __shared__ float qlds[128];
  __shared__ float s_lds[1024];
  __shared__ float pool[64];
  __shared__ float red[256];
  __shared__ float clds[128];
  __shared__ float pv[16][16][8];   // pooled_v2 partials: [slice][chunk][elem]

  const int tid  = threadIdx.x;
  const int lane = tid & 31;
  const int wave = tid >> 5;
  const int ml   = lane & 15;
  const int kg   = lane >> 4;
  const int b = blockIdx.x >> 3;
  const int h = blockIdx.x & 7;

  if (tid < 128) qlds[tid] = qF[(size_t)b * DDIM + h * HDIM + tid];
  for (int m = tid; m < 1024; m += 256) s_lds[m] = bl[0];
  if (tid < 64) pool[tid] = 0.f;

  // hoist all 16 Wb B-fragments (loop-invariant over M)
  v16bf bfr[4][4];
#pragma unroll
  for (int kc = 0; kc < 4; ++kc) {
#pragma unroll
    for (int t = 0; t < 4; ++t) {
      const v8bf* br = (const v8bf*)(Wbt + (t * 16 + ml) * HDIM + kc * 32);
      v8bf lo = br[kg], hi = br[2 + kg];
      v16bf f;
#pragma unroll
      for (int i = 0; i < 8; ++i) { f[i] = lo[i]; f[i + 8] = hi[i]; }
      bfr[kc][t] = f;
    }
  }
  __syncthreads();

  for (int iter = 0; iter < 8; ++iter) {
    const int m0 = iter * 128 + wave * 16;          // this wave's 16 positions
    const int mA = m0 + ml;                         // A-row for this lane
    const size_t krow = ((size_t)(b * 1024 + mA)) * DDIM + h * HDIM;
    v8f acc[4] = {};
#pragma unroll
    for (int kc = 0; kc < 4; ++kc) {
      const int kk = kc * 32;
      const v8bf* kr = (const v8bf*)(kB + krow + kk);
      v8bf klo = kr[kg];
      v8bf khi = kr[2 + kg];
      v16bf a;
      const int k0 = kk + kg * 8, k1 = kk + 16 + kg * 8;
#pragma unroll
      for (int i = 0; i < 8; ++i) {
        a[i]     = (bf16_t)((float)klo[i] * qlds[k0 + i]);
        a[i + 8] = (bf16_t)((float)khi[i] * qlds[k1 + i]);
      }
#pragma unroll
      for (int t = 0; t < 4; ++t) {
        acc[t] = __builtin_amdgcn_wmma_f32_16x16x32_bf16(
            false, a, false, bfr[kc][t], (short)0, acc[t], false, false);
      }
    }
    // consume am immediately: spatial logits + masked pool via LDS atomics
#pragma unroll
    for (int t = 0; t < 4; ++t) {
#pragma unroll
      for (int jj = 0; jj < 8; ++jj) {
        int mg = m0 + jj + kg * 8;                  // C/D row -> position
        int n  = t * 16 + ml;                       // C/D col -> M1 index
        float am = acc[t][jj] + bb[n];
        am = am > 0.f ? am : 0.f;
        float mv = mask[(size_t)b * 1024 + mg];
        atomicAdd(&s_lds[mg], am * Wl[n]);
        atomicAdd(&pool[n], am * mv);
      }
    }
  }
  __syncthreads();

  // masked softmax over M + masksum
  float lmax = -3.402823e38f, lmsk = 0.f;
  for (int m = tid; m < 1024; m += 256) {
    float mv = mask[(size_t)b * 1024 + m];
    float sv = s_lds[m];
    sv = (mv == 0.f) ? -1e9f : sv;
    s_lds[m] = sv;
    lmax = fmaxf(lmax, sv);
    lmsk += mv;
  }
  red[tid] = lmax; __syncthreads();
  for (int o = 128; o > 0; o >>= 1) {
    if (tid < o) red[tid] = fmaxf(red[tid], red[tid + o]);
    __syncthreads();
  }
  float mx = red[0]; __syncthreads();
  red[tid] = lmsk; __syncthreads();
  for (int o = 128; o > 0; o >>= 1) {
    if (tid < o) red[tid] += red[tid + o];
    __syncthreads();
  }
  float masksum = red[0]; __syncthreads();
  float lsum = 0.f;
  for (int m = tid; m < 1024; m += 256) {
    float e = expf(s_lds[m] - mx);
    s_lds[m] = e; lsum += e;
  }
  red[tid] = lsum; __syncthreads();
  for (int o = 128; o > 0; o >>= 1) {
    if (tid < o) red[tid] += red[tid + o];
    __syncthreads();
  }
  float inv = 1.f / red[0]; __syncthreads();
  for (int m = tid; m < 1024; m += 256) s_lds[m] *= inv;

  // channel gate: sigmoid((pool/masksum) @ Wl2 + bl2)
  if (tid < 128) {
    float a = 0.f, invms = 1.f / masksum;
    for (int n = 0; n < 64; ++n) a += pool[n] * invms * Wl2[n * HDIM + tid];
    a += bl2[tid];
    clds[tid] = 1.f / (1.f + expf(-a));
  }
  __syncthreads();

  // pooled_v2 = sum_m alpha[m] * v2[b,h,m,:], vectorized 16B bf16 loads
  {
    const int c = tid & 15;        // d-chunk: d = c*8 .. c*8+7
    const int s = tid >> 4;        // m-slice: m = s, s+16, ...
    float a8[8] = {};
    for (int m = s; m < 1024; m += 16) {
      float alpha = s_lds[m];
      v8bf kv = *(const v8bf*)(v2B + ((size_t)(b * 1024 + m)) * DDIM +
                               h * HDIM + c * 8);
#pragma unroll
      for (int e = 0; e < 8; ++e) a8[e] += alpha * (float)kv[e];
    }
#pragma unroll
    for (int e = 0; e < 8; ++e) pv[s][c][e] = a8[e];
  }
  __syncthreads();
  if (tid < 128) {
    int d = tid, c = d >> 3, e = d & 7;
    float pooled = 0.f;
#pragma unroll
    for (int s = 0; s < 16; ++s) pooled += pv[s][c][e];
    size_t o = (size_t)b * DDIM + h * HDIM + d;
    out[o] = v1F[o] * pooled * clds[d];
  }
}

// ---------------------------------------------------------------------------
extern "C" void kernel_launch(void* const* d_in, const int* in_sizes, int n_in,
                              void* d_out, int out_size, void* d_ws, size_t ws_size,
                              hipStream_t stream)
{
  (void)in_sizes; (void)n_in; (void)out_size; (void)ws_size;
  const float* query  = (const float*)d_in[0];
  const float* key    = (const float*)d_in[1];
  const float* mask   = (const float*)d_in[2];
  const float* value1 = (const float*)d_in[3];
  const float* value2 = (const float*)d_in[4];
  const float* Wq  = (const float*)d_in[5];  const float* bq   = (const float*)d_in[6];
  const float* gq  = (const float*)d_in[7];  const float* gbq  = (const float*)d_in[8];
  const float* Wk  = (const float*)d_in[9];  const float* bk   = (const float*)d_in[10];
  const float* gk  = (const float*)d_in[11]; const float* gbk  = (const float*)d_in[12];
  const float* Wv1 = (const float*)d_in[13]; const float* bv1  = (const float*)d_in[14];
  const float* gv1 = (const float*)d_in[15]; const float* gbv1 = (const float*)d_in[16];
  const float* Wv2 = (const float*)d_in[17]; const float* bv2  = (const float*)d_in[18];
  const float* gv2 = (const float*)d_in[19]; const float* gbv2 = (const float*)d_in[20];
  const float* Wb  = (const float*)d_in[21]; const float* bb   = (const float*)d_in[22];
  const float* Wl  = (const float*)d_in[23]; const float* bl   = (const float*)d_in[24];
  const float* Wl2 = (const float*)d_in[25]; const float* bl2  = (const float*)d_in[26];
  float* out = (float*)d_out;

  char* ws = (char*)d_ws;
  size_t off = 0;
  auto alloc = [&](size_t bytes) -> char* {
    char* p = ws + off;
    off += (bytes + 255) & ~(size_t)255;
    return p;
  };
  bf16_t* WtQ   = (bf16_t*)alloc((size_t)1024 * 1024 * 2);
  bf16_t* WtK   = (bf16_t*)alloc((size_t)1024 * 1024 * 2);
  bf16_t* WtV1  = (bf16_t*)alloc((size_t)1024 * 1024 * 2);
  bf16_t* WtV2  = (bf16_t*)alloc((size_t)1024 * 1024 * 2);
  bf16_t* Wbt   = (bf16_t*)alloc((size_t)64 * 128 * 2);
  float*  qBuf  = (float*)alloc((size_t)64 * 1024 * 4);
  float*  v1Buf = (float*)alloc((size_t)64 * 1024 * 4);
  bf16_t* kBuf  = (bf16_t*)alloc((size_t)64 * 1024 * 1024 * 2);
  bf16_t* v2Buf = (bf16_t*)alloc((size_t)64 * 1024 * 1024 * 2);

  {
    int total = 1024 * 1024, thr = 256, blk = (total + thr - 1) / thr;
    convert_transpose<<<blk, thr, 0, stream>>>(Wq,  WtQ,  1024, 1024);
    convert_transpose<<<blk, thr, 0, stream>>>(Wk,  WtK,  1024, 1024);
    convert_transpose<<<blk, thr, 0, stream>>>(Wv1, WtV1, 1024, 1024);
    convert_transpose<<<blk, thr, 0, stream>>>(Wv2, WtV2, 1024, 1024);
    convert_transpose<<<(128 * 64 + 255) / 256, 256, 0, stream>>>(Wb, Wbt, 128, 64);
  }

  // q / v1: 64 rows -> 4 blocks; k / v2: 65536 rows -> 4096 blocks
  branch_kernel<<<4,    256, 0, stream>>>(query,  WtQ,  bq,  gq,  gbq,  qBuf,  nullptr);
  branch_kernel<<<4,    256, 0, stream>>>(value1, WtV1, bv1, gv1, gbv1, v1Buf, nullptr);
  branch_kernel<<<4096, 256, 0, stream>>>(key,    WtK,  bk,  gk,  gbk,  nullptr, kBuf);
  branch_kernel<<<4096, 256, 0, stream>>>(value2, WtV2, bv2, gv2, gbv2, nullptr, v2Buf);

  attn_kernel<<<512, 256, 0, stream>>>(qBuf, kBuf, v1Buf, v2Buf, mask,
                                       Wbt, bb, Wl, bl, Wl2, bl2, out);
}
